// PhysicalNeuralNetwork_15152644620825
// MI455X (gfx1250) — matrix operand order repaired and checked by reading, not compiled, not run on hardware
//
#include <hip/hip_runtime.h>

typedef __attribute__((ext_vector_type(16))) _Float16 v16h;
typedef __attribute__((ext_vector_type(8)))  _Float16 v8h;
typedef __attribute__((ext_vector_type(8)))  float    v8f;

#define IN_SZ   64
#define HID_SZ  128
#define OUT_SZ  64
#define BM      128          // rows of x per block (8 waves x 16 rows)
#define THREADS 256
#define NBLOCKS 8192         // 1048576 / BM

union FragAB { v16h v; v8h h[2]; };

__device__ __forceinline__ v8h cvt8(float4 a, float4 b) {
  v8h r = { (_Float16)a.x, (_Float16)a.y, (_Float16)a.z, (_Float16)a.w,
            (_Float16)b.x, (_Float16)b.y, (_Float16)b.z, (_Float16)b.w };
  return r;
}

// sigmoid(v) = 0.5 * tanh(0.5*v) + 0.5   (exact identity)
// CDNA5 has hardware V_TANH_F32 (TRANS op): 1 trans + 2 fma per element,
// vs the libm path's exp + rcp (2 trans) + 3 valu.
// Caller passes y = 0.5*v (lets the bias-add fold into one FMA upstream).
__device__ __forceinline__ float half_tanh_half(float y) {
  float t;
#if defined(__HIP_DEVICE_COMPILE__)
#if __has_builtin(__builtin_amdgcn_tanhf)
  t = __builtin_amdgcn_tanhf(y);
#else
  // v_nop satisfies the TRANS-op hazard (1 independent op before use);
  // the compiler can't insert delays around opaque asm.
  asm("v_tanh_f32 %0, %1\n\tv_nop" : "=v"(t) : "v"(y));
#endif
#else
  t = y; // host pass: never executed
#endif
  return __builtin_fmaf(0.5f, t, 0.5f);
}

// ---------------------------------------------------------------------------
// One-time weight "physics": fold trace resistance into the weights, emit
// f16 *transposed* copies (N-major) so WMMA B-fragments are contiguous.
// w1t: [H][I] = W1_eff^T ; w2t: [O][H] = W2_eff^T
// ---------------------------------------------------------------------------
__global__ __launch_bounds__(THREADS) void prep_weights(
    const float* __restrict__ W1, const float* __restrict__ W2,
    const float* __restrict__ ip, const float* __restrict__ hp,
    const float* __restrict__ op,
    _Float16* __restrict__ w1t, _Float16* __restrict__ w2t)
{
  const float RHO_OVER_T = 1.68e-8f / 35e-6f;   // 4.8e-4
  int idx = blockIdx.x * blockDim.x + threadIdx.x;
  if (idx < IN_SZ * HID_SZ) {
    int h = idx / IN_SZ, i = idx % IN_SZ;
    float L = fabsf(ip[2*i]   - hp[2*h]) +
              fabsf(ip[2*i+1] - hp[2*h+1]);
    float w = W1[i * HID_SZ + h];
    float norm  = (fminf(fmaxf(w, -3.0f), 3.0f) + 3.0f) * (1.0f / 6.0f);
    float width = 0.09f + norm * (5.0f - 0.09f);
    float R = RHO_OVER_T * L / width;
    w1t[idx] = (_Float16)(w / (R + 1.0f));
  } else if (idx < IN_SZ * HID_SZ + HID_SZ * OUT_SZ) {
    int j = idx - IN_SZ * HID_SZ;
    int o = j / HID_SZ, h = j % HID_SZ;
    float L = fabsf(hp[2*h]   - op[2*o]) +
              fabsf(hp[2*h+1] - op[2*o+1]);
    float w = W2[h * OUT_SZ + o];
    float norm  = (fminf(fmaxf(w, -3.0f), 3.0f) + 3.0f) * (1.0f / 6.0f);
    float width = 0.09f + norm * (5.0f - 0.09f);
    float R = RHO_OVER_T * L / width;
    w2t[j] = (_Float16)(w / (R + 1.0f));
  }
}

// ---------------------------------------------------------------------------
// Fused 2-layer MLP, f16 WMMA with f32 accumulation.
// Each wave: 16 rows. GEMM1: 2 K-steps x 8 N-tiles. GEMM2: 4 K-steps x 4 N-tiles.
// Hidden tile staged in LDS (f16) to re-layout C-fragments into A-fragments.
// ---------------------------------------------------------------------------
__global__ __launch_bounds__(THREADS) void mlp_wmma(
    const float* __restrict__ x,
    const float* __restrict__ b1, const float* __restrict__ b2,
    const _Float16* __restrict__ w1t, const _Float16* __restrict__ w2t,
    float* __restrict__ out)
{
  __shared__ _Float16 sW[HID_SZ * IN_SZ];   // 8192 halves (16 KB), reused W1t->W2t
  __shared__ _Float16 sH[8 * 16 * HID_SZ];  // 16384 halves (32 KB) hidden staging

  const int tid  = threadIdx.x;
  const int wave = tid >> 5;
  const int lane = tid & 31;
  const int lrow = lane & 15;   // M (A) or N (B/C) within a 16-tile
  const int hi   = lane >> 4;   // lane-group select
  const size_t row0 = (size_t)blockIdx.x * BM;

  // ---- stage W1^T into LDS (1024 float4 = 16 KB, coalesced b128) ----
  {
    const float4* g = (const float4*)w1t;
    float4* s = (float4*)sW;
    #pragma unroll
    for (int k = 0; k < 4; ++k) s[tid + k * THREADS] = g[tid + k * THREADS];
  }
  __syncthreads();

  // ---- GEMM1: hidden(16x128) = x(16x64) @ W1_eff(64x128) ----
  const float* xrow = x + (row0 + (size_t)(wave * 16 + lrow)) * IN_SZ;
  v8f acc[8] = {};
  #pragma unroll
  for (int ks = 0; ks < 2; ++ks) {
    // A-fragment straight from global, f32 -> f16 in registers.
    // K layout: e=0..7 -> K = ks*32 + hi*8 + e ; e=8..15 -> +16
    const float* ap = xrow + ks * 32 + hi * 8;
    float4 f0 = *(const float4*)(ap + 0);
    float4 f1 = *(const float4*)(ap + 4);
    float4 f2 = *(const float4*)(ap + 16);
    float4 f3 = *(const float4*)(ap + 20);
    FragAB a; a.h[0] = cvt8(f0, f1); a.h[1] = cvt8(f2, f3);
    #pragma unroll
    for (int n = 0; n < 8; ++n) {
      // B-fragment: col N = n*16+lrow, K = ks*32 + hi*8 + e ; e=8..15 -> +16
      const _Float16* bp = sW + (n * 16 + lrow) * IN_SZ + ks * 32 + hi * 8;
      FragAB b;
      b.h[0] = *(const v8h*)(bp);
      b.h[1] = *(const v8h*)(bp + 16);
      acc[n] = __builtin_amdgcn_wmma_f32_16x16x32_f16(
          false, a.v, false, b.v, (short)0, acc[n], false, false);
    }
  }

  // ---- epilogue 1: sigmoid(acc + b1) via hw tanh -> f16 staging in LDS ----
  //      (C layout: M = r + hi*8, N = lrow)
  _Float16* hbase = sH + wave * (16 * HID_SZ);
  #pragma unroll
  for (int n = 0; n < 8; ++n) {
    float hb = 0.5f * b1[n * 16 + lrow];   // fold bias into the 0.5*v scale
    #pragma unroll
    for (int r = 0; r < 8; ++r) {
      float y = __builtin_fmaf(acc[n][r], 0.5f, hb);
      float v = half_tanh_half(y);
      hbase[(r + hi * 8) * HID_SZ + n * 16 + lrow] = (_Float16)v;
    }
  }
  __syncthreads();   // all waves done reading sW (W1)

  // ---- stage W2^T into the same LDS buffer ----
  {
    const float4* g = (const float4*)w2t;
    float4* s = (float4*)sW;
    #pragma unroll
    for (int k = 0; k < 4; ++k) s[tid + k * THREADS] = g[tid + k * THREADS];
  }
  __syncthreads();

  // ---- GEMM2: out(16x64) = hidden(16x128) @ W2_eff(128x64) ----
  v8f acc2[4] = {};
  #pragma unroll
  for (int ks = 0; ks < 4; ++ks) {
    const _Float16* ap = hbase + lrow * HID_SZ + ks * 32 + hi * 8;
    FragAB a;
    a.h[0] = *(const v8h*)(ap);
    a.h[1] = *(const v8h*)(ap + 16);
    #pragma unroll
    for (int n = 0; n < 4; ++n) {
      const _Float16* bp = sW + (n * 16 + lrow) * HID_SZ + ks * 32 + hi * 8;
      FragAB b;
      b.h[0] = *(const v8h*)(bp);
      b.h[1] = *(const v8h*)(bp + 16);
      acc2[n] = __builtin_amdgcn_wmma_f32_16x16x32_f16(
          false, a.v, false, b.v, (short)0, acc2[n], false, false);
    }
  }

  // ---- epilogue 2: sigmoid via hw tanh -> global f32 (64B/row-group stores) ----
  float* orow = out + (row0 + (size_t)(wave * 16)) * OUT_SZ;
  #pragma unroll
  for (int n = 0; n < 4; ++n) {
    float hb = 0.5f * b2[n * 16 + lrow];
    #pragma unroll
    for (int r = 0; r < 8; ++r) {
      float y = __builtin_fmaf(acc2[n][r], 0.5f, hb);
      float v = half_tanh_half(y);
      orow[(size_t)(r + hi * 8) * OUT_SZ + n * 16 + lrow] = v;
    }
  }
}

// ---------------------------------------------------------------------------
extern "C" void kernel_launch(void* const* d_in, const int* in_sizes, int n_in,
                              void* d_out, int out_size, void* d_ws, size_t ws_size,
                              hipStream_t stream) {
  (void)in_sizes; (void)n_in; (void)out_size; (void)ws_size;
  const float* x  = (const float*)d_in[0];
  const float* W1 = (const float*)d_in[1];
  const float* b1 = (const float*)d_in[2];
  const float* W2 = (const float*)d_in[3];
  const float* b2 = (const float*)d_in[4];
  const float* ip = (const float*)d_in[5];
  const float* hp = (const float*)d_in[6];
  const float* op = (const float*)d_in[7];

  _Float16* w1t = (_Float16*)d_ws;                 // 8192 halves
  _Float16* w2t = w1t + IN_SZ * HID_SZ;            // 8192 halves (32 KB total)

  int prep_elems = IN_SZ * HID_SZ + HID_SZ * OUT_SZ;
  prep_weights<<<(prep_elems + THREADS - 1) / THREADS, THREADS, 0, stream>>>(
      W1, W2, ip, hp, op, w1t, w2t);

  mlp_wmma<<<NBLOCKS, THREADS, 0, stream>>>(x, b1, b2, w1t, w2t, (float*)d_out);
}
